// PointConv_50345606644345
// MI455X (gfx1250) — matrix operand order repaired and checked by reading, not compiled, not run on hardware
//
#include <hip/hip_runtime.h>

typedef __attribute__((ext_vector_type(16))) _Float16 v16h;
typedef __attribute__((ext_vector_type(8)))  _Float16 v8h;
typedef __attribute__((ext_vector_type(8)))  float    v8f;

#define B_    8
#define N_    8192
#define S_    2048
#define K_    32
#define P_    (B_ * S_)            // 16384 centroids
#define COLS_ ((size_t)P_ * K_)    // 524288 sample positions
#define D_    64
#define EPS_  1e-5f

// B-matrix fragment swizzle (this file's contiguous k order, f16 32x16 B tile, wave32):
// element (k, col) of X[Kd, Nc] -> lane = (k%32/16)*16 + col%16, e = k%16,
// fragments stored contiguously per (ktile, ntile, lane): 16 halves = 32B.
__device__ __forceinline__ size_t bswz(int k, int col, int NT) {
  int kt = k >> 5, kl = k & 31;
  int lane = ((kl >> 4) << 4) | (col & 15);
  int e = kl & 15;
  return ((((size_t)kt * NT + (col >> 4)) << 5) + (size_t)lane) * 16 + e;
}

// ---------------------------------------------------------------------------
// 1. Farthest point sampling: one persistent block per batch (serial chain).
// ---------------------------------------------------------------------------
__global__ __launch_bounds__(1024) void fps_kernel(
    const float* __restrict__ xyz,   // [B,3,N]
    int* __restrict__ fidx,          // [B,S]
    float* __restrict__ nxyz,        // [P,3]
    float* __restrict__ out_xyz)     // d_out part 1: [B,3,S]
{
  const int b = blockIdx.x, tid = threadIdx.x;
  __shared__ float dist[N_];
  __shared__ float rv[1024];
  __shared__ int   ri[1024];
  __shared__ float cen[3];
  const float* px = xyz + (size_t)b * 3 * N_;
  for (int j = tid; j < N_; j += 1024) dist[j] = 1e10f;
  int far = 0;
  __syncthreads();
  for (int s = 0; s < S_; ++s) {
    if (tid == 0) {
      float cx = px[far], cy = px[N_ + far], cz = px[2 * N_ + far];
      cen[0] = cx; cen[1] = cy; cen[2] = cz;
      fidx[b * S_ + s] = far;
      size_t p = (size_t)b * S_ + s;
      nxyz[p * 3 + 0] = cx; nxyz[p * 3 + 1] = cy; nxyz[p * 3 + 2] = cz;
      out_xyz[(size_t)b * 3 * S_ + 0 * S_ + s] = cx;
      out_xyz[(size_t)b * 3 * S_ + 1 * S_ + s] = cy;
      out_xyz[(size_t)b * 3 * S_ + 2 * S_ + s] = cz;
    }
    __syncthreads();
    float cx = cen[0], cy = cen[1], cz = cen[2];
    float best = -1.f; int bi = 0;
    for (int j = tid; j < N_; j += 1024) {
      float dx = px[j] - cx, dy = px[N_ + j] - cy, dz = px[2 * N_ + j] - cz;
      float d = dx * dx + dy * dy + dz * dz;
      float dd = fminf(dist[j], d);
      dist[j] = dd;
      if (dd > best) { best = dd; bi = j; }
    }
    rv[tid] = best; ri[tid] = bi;
    __syncthreads();
    for (int o = 512; o > 0; o >>= 1) {
      if (tid < o) {
        float a = rv[tid], c = rv[tid + o];
        if (c > a || (c == a && ri[tid + o] < ri[tid])) { rv[tid] = c; ri[tid] = ri[tid + o]; }
      }
      __syncthreads();
    }
    far = ri[0];
    __syncthreads();
  }
}

// ---------------------------------------------------------------------------
// 2. KNN: one block per centroid; dist[8192] in LDS, 32 iterated argmins.
// ---------------------------------------------------------------------------
__global__ __launch_bounds__(256) void knn_kernel(
    const float* __restrict__ xyz, const float* __restrict__ nxyz,
    int* __restrict__ knn)
{
  const int p = blockIdx.x, b = p >> 11, tid = threadIdx.x;
  __shared__ float dist[N_];
  __shared__ float rv[256];
  __shared__ int   ri[256];
  const float* px = xyz + (size_t)b * 3 * N_;
  float cx = nxyz[(size_t)p * 3 + 0], cy = nxyz[(size_t)p * 3 + 1], cz = nxyz[(size_t)p * 3 + 2];
  for (int j = tid; j < N_; j += 256) {
    float dx = px[j] - cx, dy = px[N_ + j] - cy, dz = px[2 * N_ + j] - cz;
    dist[j] = dx * dx + dy * dy + dz * dz;
  }
  __syncthreads();
  for (int kk = 0; kk < K_; ++kk) {
    float best = 3.4e38f; int bi = 0;
    for (int j = tid; j < N_; j += 256) {
      float d = dist[j];
      if (d < best) { best = d; bi = j; }
    }
    rv[tid] = best; ri[tid] = bi;
    __syncthreads();
    for (int o = 128; o > 0; o >>= 1) {
      if (tid < o) {
        float a = rv[tid], c = rv[tid + o];
        if (c < a || (c == a && ri[tid + o] < ri[tid])) { rv[tid] = c; ri[tid] = ri[tid + o]; }
      }
      __syncthreads();
    }
    if (tid == 0) { knn[(size_t)p * K_ + kk] = ri[0]; dist[ri[0]] = 3.4e38f; }
    __syncthreads();
  }
}

// ---------------------------------------------------------------------------
// 3. Gather: build npts directly in B-fragment-swizzled f16 layout
//    ([96 x COLS] logical; rows 0-2 gnorm, 3-66 points, 67-95 zero pad),
//    plus gnorm (f32, [3 x COLS]) for the weight branch.
// ---------------------------------------------------------------------------
__global__ __launch_bounds__(256) void gather_kernel(
    const float* __restrict__ xyz, const float* __restrict__ points,
    const int* __restrict__ knn, const float* __restrict__ nxyz,
    _Float16* __restrict__ nptsH, float* __restrict__ gnormF)
{
  size_t t = (size_t)blockIdx.x * 256 + threadIdx.x;
  if (t >= COLS_) return;
  const int NT = (int)(COLS_ >> 4);
  int col = (int)t;
  int p = (int)(t >> 5), b = p >> 11;
  int j = knn[t];
  const float* px = xyz + (size_t)b * 3 * N_;
  float gx = px[j]          - nxyz[(size_t)p * 3 + 0];
  float gy = px[N_ + j]     - nxyz[(size_t)p * 3 + 1];
  float gz = px[2 * N_ + j] - nxyz[(size_t)p * 3 + 2];
  nptsH[bswz(0, col, NT)] = (_Float16)gx;
  nptsH[bswz(1, col, NT)] = (_Float16)gy;
  nptsH[bswz(2, col, NT)] = (_Float16)gz;
  gnormF[0 * COLS_ + t] = gx; gnormF[1 * COLS_ + t] = gy; gnormF[2 * COLS_ + t] = gz;
  const float* pp = points + (size_t)b * D_ * N_;
  for (int d = 0; d < D_; ++d)
    nptsH[bswz(3 + d, col, NT)] = (_Float16)pp[(size_t)d * N_ + j];
  for (int r = 3 + D_; r < 96; ++r)
    nptsH[bswz(r, col, NT)] = (_Float16)0.f;
}

// ---------------------------------------------------------------------------
// 4. Weight pack: fp32 [M,Ks] -> f16 A-fragment-swizzled [M/16, KT, 32, 16]
//    (contiguous k = 16*half + e, same order as bswz; each lane fragment contiguous 32B).
// ---------------------------------------------------------------------------
__global__ void pack_w_kernel(const float* __restrict__ src, _Float16* __restrict__ dst,
                              int M, int Ks, int Kd)
{
  int KT = Kd >> 5;
  int idx = blockIdx.x * 256 + threadIdx.x;
  if (idx >= M * Kd) return;
  int tile = idx >> 9;           // (mt, kt) block of 512 halves
  int within = idx & 511;
  int lane = within >> 4, e = within & 15;
  int mt = tile / KT, kt = tile % KT;
  int l16 = lane & 15, half_ = lane >> 4;
  int kl = half_ * 16 + e;
  int m = mt * 16 + l16;
  int k = kt * 32 + kl;
  dst[idx] = (k < Ks) ? (_Float16)src[m * Ks + k] : (_Float16)0.f;
}

// ---------------------------------------------------------------------------
// 5. WMMA GEMM: Y[M,Nc] = W[M,Kd] * X[Kd,Nc], fragment-swizzled f16 inputs.
//    Each wave computes a 32x64 tile: 2 M-tiles x 4 N-tiles -> 8 WMMAs/K-step
//    fed by 6 contiguous 32B fragment loads (A reused 4x, B reused 2x).
// ---------------------------------------------------------------------------
__global__ __launch_bounds__(256) void wmma_gemm_kernel(
    const _Float16* __restrict__ Ws,  // A swizzled: (M/16)*KT*512 halves
    const _Float16* __restrict__ Xs,  // B swizzled: KT*NT*512 halves
    float* __restrict__ Y, int M, int Kd, int Nc)
{
  const int NT = Nc >> 4, KT = Kd >> 5;
  int wave = (blockIdx.x * blockDim.x + threadIdx.x) >> 5;
  int lane = threadIdx.x & 31;
  int ngroups = NT >> 2;
  int mg = wave / ngroups, ng = wave % ngroups;
  if (mg * 32 >= M) return;
  int mt0 = mg * 2, nt0 = ng * 4;
  const v16h* Ap = (const v16h*)Ws;
  const v16h* Bp = (const v16h*)Xs;
  v8f acc[8] = {};
  for (int kt = 0; kt < KT; ++kt) {
    if (kt + 1 < KT)
      __builtin_prefetch(&Bp[((size_t)(kt + 1) * NT + nt0) * 32 + lane], 0, 0);
    v16h a0 = Ap[((size_t)(mt0 + 0) * KT + kt) * 32 + lane];
    v16h a1 = Ap[((size_t)(mt0 + 1) * KT + kt) * 32 + lane];
    v16h b0 = Bp[((size_t)kt * NT + nt0 + 0) * 32 + lane];
    v16h b1 = Bp[((size_t)kt * NT + nt0 + 1) * 32 + lane];
    v16h b2 = Bp[((size_t)kt * NT + nt0 + 2) * 32 + lane];
    v16h b3 = Bp[((size_t)kt * NT + nt0 + 3) * 32 + lane];
    acc[0] = __builtin_amdgcn_wmma_f32_16x16x32_f16(false, a0, false, b0, (short)0, acc[0], false, false);
    acc[1] = __builtin_amdgcn_wmma_f32_16x16x32_f16(false, a0, false, b1, (short)0, acc[1], false, false);
    acc[2] = __builtin_amdgcn_wmma_f32_16x16x32_f16(false, a0, false, b2, (short)0, acc[2], false, false);
    acc[3] = __builtin_amdgcn_wmma_f32_16x16x32_f16(false, a0, false, b3, (short)0, acc[3], false, false);
    acc[4] = __builtin_amdgcn_wmma_f32_16x16x32_f16(false, a1, false, b0, (short)0, acc[4], false, false);
    acc[5] = __builtin_amdgcn_wmma_f32_16x16x32_f16(false, a1, false, b1, (short)0, acc[5], false, false);
    acc[6] = __builtin_amdgcn_wmma_f32_16x16x32_f16(false, a1, false, b2, (short)0, acc[6], false, false);
    acc[7] = __builtin_amdgcn_wmma_f32_16x16x32_f16(false, a1, false, b3, (short)0, acc[7], false, false);
  }
  int l16 = lane & 15, half_ = lane >> 4;
#pragma unroll
  for (int mi = 0; mi < 2; ++mi)
#pragma unroll
    for (int t = 0; t < 4; ++t)
#pragma unroll
      for (int v = 0; v < 8; ++v) {
        int m = (mt0 + mi) * 16 + v + half_ * 8;
        int col = (nt0 + t) * 16 + l16;
        Y[(size_t)m * Nc + col] = acc[mi * 4 + t][v];
      }
}

// ---------------------------------------------------------------------------
// 6. BN stats: per-channel sum / sumsq (LDS tree + float atomics).
// ---------------------------------------------------------------------------
__global__ __launch_bounds__(256) void zero_kernel(float* p, int n)
{
  int i = blockIdx.x * 256 + threadIdx.x;
  if (i < n) p[i] = 0.f;
}

__global__ __launch_bounds__(256) void bn_stats_kernel(
    const float* __restrict__ Y, float* __restrict__ stats, unsigned long long Nc)
{
  int m = blockIdx.x;
  __shared__ float ssum[256], ssq[256];
  float s = 0.f, q = 0.f;
  for (size_t i = (size_t)blockIdx.y * 256 + threadIdx.x; i < Nc;
       i += (size_t)gridDim.y * 256) {
    float v = Y[(size_t)m * Nc + i];
    s += v; q += v * v;
  }
  ssum[threadIdx.x] = s; ssq[threadIdx.x] = q;
  __syncthreads();
  for (int o = 128; o > 0; o >>= 1) {
    if (threadIdx.x < o) {
      ssum[threadIdx.x] += ssum[threadIdx.x + o];
      ssq[threadIdx.x]  += ssq[threadIdx.x + o];
    }
    __syncthreads();
  }
  if (threadIdx.x == 0) {
    atomicAdd(&stats[2 * m], ssum[0]);
    atomicAdd(&stats[2 * m + 1], ssq[0]);
  }
}

// ---------------------------------------------------------------------------
// 7. BN apply + ReLU. Optional f16 out (row-major or B-fragment-swizzled)
//    and/or f32 out. Biases cancel exactly under train-mode BN.
// ---------------------------------------------------------------------------
__global__ __launch_bounds__(256) void bn_apply_kernel(
    const float* __restrict__ Y, const float* __restrict__ stats,
    const float* __restrict__ g, const float* __restrict__ be,
    unsigned long long Nc, _Float16* outH, float* outF, int swz)
{
  int m = blockIdx.y;
  size_t i = (size_t)blockIdx.x * 256 + threadIdx.x;
  if (i >= Nc) return;
  float inv  = 1.0f / (float)Nc;
  float mean = stats[2 * m] * inv;
  float var  = stats[2 * m + 1] * inv - mean * mean;
  float v = Y[(size_t)m * Nc + i];
  v = fmaxf((v - mean) * rsqrtf(var + EPS_) * g[m] + be[m], 0.f);
  if (outH) {
    if (swz) outH[bswz(m, (int)i, (int)(Nc >> 4))] = (_Float16)v;
    else     outH[(size_t)m * Nc + i] = (_Float16)v;
  }
  if (outF) outF[(size_t)m * Nc + i] = v;
}

// Final BN -> d_out part 2, scattering [c, b*S+s] -> [b,128,S].
__global__ __launch_bounds__(256) void bn_apply_out_kernel(
    const float* __restrict__ Y, const float* __restrict__ stats,
    const float* __restrict__ g, const float* __restrict__ be,
    float* __restrict__ out)
{
  int m = blockIdx.y;
  size_t i = (size_t)blockIdx.x * 256 + threadIdx.x;
  if (i >= (size_t)P_) return;
  float inv  = 1.0f / (float)P_;
  float mean = stats[2 * m] * inv;
  float var  = stats[2 * m + 1] * inv - mean * mean;
  float v = Y[(size_t)m * P_ + i];
  v = fmaxf((v - mean) * rsqrtf(var + EPS_) * g[m] + be[m], 0.f);
  int b = (int)(i >> 11), s = (int)(i & 2047);
  out[(size_t)b * 128 * S_ + (size_t)m * S_ + s] = v;
}

// ---------------------------------------------------------------------------
// 8. Small VALU conv for the 3->8->8->16 weight branch (tiny FLOPs).
// ---------------------------------------------------------------------------
__global__ __launch_bounds__(256) void small_conv_kernel(
    const float* __restrict__ in, const float* __restrict__ W,
    float* __restrict__ out, int Cin, int Cout)
{
  size_t col = (size_t)blockIdx.x * 256 + threadIdx.x;
  if (col >= COLS_) return;
  float xin[8];
  for (int c = 0; c < Cin; ++c) xin[c] = in[(size_t)c * COLS_ + col];
  for (int o = 0; o < Cout; ++o) {
    float acc = 0.f;
    for (int c = 0; c < Cin; ++c) acc += W[o * Cin + c] * xin[c];
    out[(size_t)o * COLS_ + col] = acc;
  }
}

// ---------------------------------------------------------------------------
// 9. Per-centroid feat x w einsum: 128x16x32 batched WMMA.
//    feat A-fragment = one contiguous 32B run; wf B-fragment = one 32B run.
//    Output written directly in the final GEMM's B-swizzled layout (NT=1024).
// ---------------------------------------------------------------------------
__global__ __launch_bounds__(256) void pcw_gemm_kernel(
    const _Float16* __restrict__ feat,   // [128, COLS] row-major f16
    const _Float16* __restrict__ wf,     // [16, COLS] row-major f16
    _Float16* __restrict__ X3)           // B-swizzled [2048, P]
{
  int wave = (blockIdx.x * blockDim.x + threadIdx.x) >> 5;
  int lane = threadIdx.x & 31;
  int p = wave >> 3, mt = wave & 7;
  if (p >= P_) return;
  int half_ = lane >> 4, l16 = lane & 15;
  size_t base = (size_t)p * K_;
  const _Float16* ap = feat + (size_t)(mt * 16 + l16) * COLS_ + base + half_ * 16;
  v16h a = *(const v16h*)(ap);         // k = base + half*16 + 0..15
  v16h b = *(const v16h*)(wf + (size_t)l16 * COLS_ + base + half_ * 16);
  v8f acc = {};
  acc = __builtin_amdgcn_wmma_f32_16x16x32_f16(false, a, false, b,
                                               (short)0, acc, false, false);
#pragma unroll
  for (int v = 0; v < 8; ++v) {
    int c = mt * 16 + v + half_ * 8;     // feat channel
    int r = c * 16 + l16;                // reshape row index (c*16 + wch)
    X3[bswz(r, p, P_ >> 4)] = (_Float16)acc[v];
  }
}

// ---------------------------------------------------------------------------
// Orchestration
// ---------------------------------------------------------------------------
extern "C" void kernel_launch(void* const* d_in, const int* in_sizes, int n_in,
                              void* d_out, int out_size, void* d_ws, size_t ws_size,
                              hipStream_t stream) {
  (void)in_sizes; (void)n_in; (void)out_size; (void)ws_size;
  const float* xyz    = (const float*)d_in[0];
  const float* points = (const float*)d_in[1];
  const float* w1_0 = (const float*)d_in[2];
  const float* g1_0 = (const float*)d_in[4];
  const float* be1_0 = (const float*)d_in[5];
  const float* w1_1 = (const float*)d_in[6];
  const float* g1_1 = (const float*)d_in[8];
  const float* be1_1 = (const float*)d_in[9];
  const float* w2_0 = (const float*)d_in[10];
  const float* g2_0 = (const float*)d_in[12];
  const float* be2_0 = (const float*)d_in[13];
  const float* w2_1 = (const float*)d_in[14];
  const float* g2_1 = (const float*)d_in[16];
  const float* be2_1 = (const float*)d_in[17];
  const float* w2_2 = (const float*)d_in[18];
  const float* g2_2 = (const float*)d_in[20];
  const float* be2_2 = (const float*)d_in[21];
  const float* lw = (const float*)d_in[22];
  const float* gl = (const float*)d_in[24];
  const float* bl = (const float*)d_in[25];
  float* outv = (float*)d_out;

  char* ws = (char*)d_ws;
  size_t off = 0;
  auto carve = [&](size_t bytes) { char* r = ws + off; off += (bytes + 255) & ~(size_t)255; return r; };
  int*      fidx    = (int*)carve((size_t)P_ * 4);
  int*      knn     = (int*)carve(COLS_ * 4);
  float*    nxyz    = (float*)carve((size_t)P_ * 3 * 4);
  _Float16* nptsH   = (_Float16*)carve(96 * COLS_ * 2);           // B-swizzled
  float*    gnormF  = (float*)carve(3 * COLS_ * 4);
  float*    Ybuf    = (float*)carve(128 * COLS_ * 4);
  _Float16* X2h     = (_Float16*)carve(64 * COLS_ * 2);           // B-swizzled
  _Float16* feath   = (_Float16*)carve(128 * COLS_ * 2);          // row-major
  float*    wbufA   = (float*)carve(16 * COLS_ * 4);
  float*    wbufB   = (float*)carve(16 * COLS_ * 4);
  _Float16* wsmallH = (_Float16*)carve(16 * COLS_ * 2);           // row-major
  _Float16* X3h     = (_Float16*)carve((size_t)2048 * P_ * 2);    // B-swizzled
  float*    Y4      = (float*)carve((size_t)128 * P_ * 4);
  _Float16* Wh1     = (_Float16*)carve((size_t)64 * 96 * 2);      // A-swizzled
  _Float16* Wh2     = (_Float16*)carve((size_t)128 * 64 * 2);     // A-swizzled
  _Float16* Wh4     = (_Float16*)carve((size_t)128 * 2048 * 2);   // A-swizzled
  float*    stats   = (float*)carve(256 * 4);

  const int cblk = (int)(COLS_ / 256);   // 2048
  const int NT_C = (int)(COLS_ >> 4);    // 32768
  const int NT_P = P_ >> 4;              // 1024

  // Stage A: FPS + new_xyz (also writes d_out part 1)
  fps_kernel<<<B_, 1024, 0, stream>>>(xyz, fidx, nxyz, outv);
  // Stage B: KNN
  knn_kernel<<<P_, 256, 0, stream>>>(xyz, nxyz, knn);
  // Stage C: gather / feature build (swizzled f16 + f32 gnorm)
  gather_kernel<<<cblk, 256, 0, stream>>>(xyz, points, knn, nxyz, nptsH, gnormF);
  // Weight packs (A-fragment swizzled)
  pack_w_kernel<<<(64 * 96 + 255) / 256, 256, 0, stream>>>(w1_0, Wh1, 64, 67, 96);
  pack_w_kernel<<<(128 * 64 + 255) / 256, 256, 0, stream>>>(w1_1, Wh2, 128, 64, 64);
  pack_w_kernel<<<(128 * 2048 + 255) / 256, 256, 0, stream>>>(lw, Wh4, 128, 2048, 2048);

  // conv1: 67->64 (WMMA), BN+ReLU -> X2h (swizzled)
  wmma_gemm_kernel<<<((64 / 32) * (NT_C / 4)) / 8, 256, 0, stream>>>(Wh1, nptsH, Ybuf, 64, 96, (int)COLS_);
  zero_kernel<<<1, 256, 0, stream>>>(stats, 256);
  bn_stats_kernel<<<dim3(64, 32), 256, 0, stream>>>(Ybuf, stats, COLS_);
  bn_apply_kernel<<<dim3(cblk, 64), 256, 0, stream>>>(Ybuf, stats, g1_0, be1_0, COLS_, X2h, nullptr, 1);

  // conv2: 64->128 (WMMA), BN+ReLU -> feath (row-major, A-side of pcw)
  wmma_gemm_kernel<<<((128 / 32) * (NT_C / 4)) / 8, 256, 0, stream>>>(Wh2, X2h, Ybuf, 128, 64, (int)COLS_);
  zero_kernel<<<1, 256, 0, stream>>>(stats, 256);
  bn_stats_kernel<<<dim3(128, 32), 256, 0, stream>>>(Ybuf, stats, COLS_);
  bn_apply_kernel<<<dim3(cblk, 128), 256, 0, stream>>>(Ybuf, stats, g1_1, be1_1, COLS_, feath, nullptr, 0);

  // Weight branch: 3->8->8->16, BN+ReLU each (VALU)
  small_conv_kernel<<<cblk, 256, 0, stream>>>(gnormF, w2_0, wbufA, 3, 8);
  zero_kernel<<<1, 256, 0, stream>>>(stats, 256);
  bn_stats_kernel<<<dim3(8, 32), 256, 0, stream>>>(wbufA, stats, COLS_);
  bn_apply_kernel<<<dim3(cblk, 8), 256, 0, stream>>>(wbufA, stats, g2_0, be2_0, COLS_, nullptr, wbufB, 0);

  small_conv_kernel<<<cblk, 256, 0, stream>>>(wbufB, w2_1, wbufA, 8, 8);
  zero_kernel<<<1, 256, 0, stream>>>(stats, 256);
  bn_stats_kernel<<<dim3(8, 32), 256, 0, stream>>>(wbufA, stats, COLS_);
  bn_apply_kernel<<<dim3(cblk, 8), 256, 0, stream>>>(wbufA, stats, g2_1, be2_1, COLS_, nullptr, wbufB, 0);

  small_conv_kernel<<<cblk, 256, 0, stream>>>(wbufB, w2_2, wbufA, 8, 16);
  zero_kernel<<<1, 256, 0, stream>>>(stats, 256);
  bn_stats_kernel<<<dim3(16, 32), 256, 0, stream>>>(wbufA, stats, COLS_);
  bn_apply_kernel<<<dim3(cblk, 16), 256, 0, stream>>>(wbufA, stats, g2_2, be2_2, COLS_, wsmallH, nullptr, 0);

  // Per-centroid einsum (WMMA) -> X3 (B-swizzled for final GEMM)
  pcw_gemm_kernel<<<(P_ * 8) / 8, 256, 0, stream>>>(feath, wsmallH, X3h);

  // Final linear 2048->128 (WMMA), BN+ReLU -> d_out part 2
  wmma_gemm_kernel<<<((128 / 32) * (NT_P / 4)) / 8, 256, 0, stream>>>(Wh4, X3h, Y4, 128, 2048, P_);
  zero_kernel<<<1, 256, 0, stream>>>(stats, 256);
  bn_stats_kernel<<<dim3(128, 8), 256, 0, stream>>>(Y4, stats, (unsigned long long)P_);
  bn_apply_out_kernel<<<dim3(P_ / 256, 128), 256, 0, stream>>>(Y4, stats, gl, bl,
                                                               outv + (size_t)B_ * 3 * S_);
}
